// RelPositionMultiHeadedAttention_73710228734782
// MI455X (gfx1250) — matrix operand order repaired and checked by reading, not compiled, not run on hardware
//
#include <hip/hip_runtime.h>
#include <hip/hip_bf16.h>
#include <stdint.h>
#include <stddef.h>

// Problem constants (match reference)
#define B_    4
#define T_    2048
#define D_    256
#define H_    4
#define DK_   64
#define P_    4095
#define PPAD_ 4112   // P padded (zero-filled) so banded reads stay in bounds

typedef __attribute__((ext_vector_type(16))) __bf16 v16bf;
typedef __attribute__((ext_vector_type(8)))  __bf16 v8bf;
typedef __attribute__((ext_vector_type(8)))  float  v8f;
typedef __attribute__((ext_vector_type(4)))  unsigned int v4u;
typedef __attribute__((ext_vector_type(8)))  int v8i;
typedef __attribute__((ext_vector_type(4)))  int v4i;

// ---------------------------------------------------------------------------
// WMMA fragment helpers (wave32, v_wmma_f32_16x16x32_bf16)
// A layout: lanes 0-15 row M=lane, elems {K0..7,K16..23}; lanes 16-31 row M,
//           elems {K8..15,K24..31}.
// B layout: lanes 0-15 col N=lane {K0..7,K16..23}; lanes 16-31 col N {K8..15,K24..31}.
// C/D:      VGPR r -> (M=r / r+8, N=lane&15).
// ---------------------------------------------------------------------------
static __device__ __forceinline__ v8f wmma_bf16(v16bf a, v16bf b, v8f c) {
  return __builtin_amdgcn_wmma_f32_16x16x32_bf16(false, a, false, b, (short)0, c, false, false);
}

static __device__ __forceinline__ v16bf load_a_f32_full(const float* __restrict__ src, int ld,
                                                        int row0, int k0) {
  int lane = threadIdx.x & 31;
  int half = lane >> 4, r = lane & 15;
  const float* p0 = src + (size_t)(row0 + r) * ld + k0 + half * 8;
  v16bf a;
#pragma unroll
  for (int e = 0; e < 8; ++e) a[e] = (__bf16)p0[e];
#pragma unroll
  for (int e = 0; e < 8; ++e) a[8 + e] = (__bf16)p0[16 + e];
  return a;
}

static __device__ __forceinline__ v16bf load_a_f32_bound(const float* __restrict__ src, int ld,
                                                         int row0, int k0, int M) {
  int lane = threadIdx.x & 31;
  int half = lane >> 4, r = lane & 15;
  int row = row0 + r;
  v16bf a;
  if (row < M) {
    const float* p0 = src + (size_t)row * ld + k0 + half * 8;
#pragma unroll
    for (int e = 0; e < 8; ++e) a[e] = (__bf16)p0[e];
#pragma unroll
    for (int e = 0; e < 8; ++e) a[8 + e] = (__bf16)p0[16 + e];
  } else {
#pragma unroll
    for (int e = 0; e < 16; ++e) a[e] = (__bf16)0.0f;
  }
  return a;
}

static __device__ __forceinline__ v16bf load_a_bf16(const __bf16* __restrict__ src, int ld,
                                                    int row0, int k0) {
  int lane = threadIdx.x & 31;
  int half = lane >> 4, r = lane & 15;
  const __bf16* p0 = src + (size_t)(row0 + r) * ld + k0 + half * 8;
  v8bf lo = *(const v8bf*)p0;
  v8bf hi = *(const v8bf*)(p0 + 16);
  v16bf a;
#pragma unroll
  for (int e = 0; e < 8; ++e) { a[e] = lo[e]; a[8 + e] = hi[e]; }
  return a;
}

// B[kk][n] sourced from row-major src[n][kk]; works for global or LDS source.
static __device__ __forceinline__ v16bf load_b_bf16(const __bf16* src, int ld, int n0, int k0) {
  int lane = threadIdx.x & 31;
  int half = lane >> 4, n = lane & 15;
  const __bf16* p = src + (size_t)(n0 + n) * ld + k0 + half * 8;
  v8bf lo = *(const v8bf*)p;
  v8bf hi = *(const v8bf*)(p + 16);
  v16bf b;
#pragma unroll
  for (int e = 0; e < 8; ++e) { b[e] = lo[e]; b[8 + e] = hi[e]; }
  return b;
}

// ---------------------------------------------------------------------------
// Small utility kernels
// ---------------------------------------------------------------------------
__global__ void cvt_f32_bf16_kernel(const float* __restrict__ src, __bf16* __restrict__ dst, int n) {
  int i = blockIdx.x * blockDim.x + threadIdx.x;
  if (i < n) dst[i] = (__bf16)src[i];
}

__global__ void zero_bf16_kernel(__bf16* __restrict__ dst, int n) {
  int i = blockIdx.x * blockDim.x + threadIdx.x;
  if (i < n) dst[i] = (__bf16)0.0f;
}

// ---------------------------------------------------------------------------
// Projection GEMM:  Y = X(M x 256, fp32) @ W^T(256 x 256, bf16) + bias
// MODE 0: Q -> q_u=Y+pos_bias_u, q_v=Y+pos_bias_v  (B,H,T,DK) bf16
// MODE 1: K -> (B,H,T,DK) bf16       MODE 2: V -> (B,H,DK,T) bf16
// MODE 3: P -> (H,PPAD,DK) bf16 (bounded rows)
// ---------------------------------------------------------------------------
template <int MODE, bool BOUND>
__global__ void proj_gemm_kernel(const float* __restrict__ X, const __bf16* __restrict__ W,
                                 const float* __restrict__ bias,
                                 const float* __restrict__ bu, const float* __restrict__ bvp,
                                 __bf16* __restrict__ out0, __bf16* __restrict__ out1, int M) {
  int n0 = blockIdx.x * 16;
  int m0 = blockIdx.y * 16;
  v8f acc = {};
#pragma unroll
  for (int k0 = 0; k0 < D_; k0 += 32) {
    v16bf a = BOUND ? load_a_f32_bound(X, D_, m0, k0, M) : load_a_f32_full(X, D_, m0, k0);
    v16bf b = load_b_bf16(W, D_, n0, k0);
    acc = wmma_bf16(a, b, acc);
  }
  int lane = threadIdx.x & 31;
  int half = lane >> 4;
  int ncol = n0 + (lane & 15);
  int h  = ncol >> 6;
  int dk = ncol & (DK_ - 1);
#pragma unroll
  for (int r = 0; r < 8; ++r) {
    int m = m0 + r + half * 8;
    if (BOUND && m >= M) continue;
    float val = acc[r];
    if (bias) val += bias[ncol];
    if (MODE == 0) {
      int b_ = m >> 11, t = m & (T_ - 1);
      size_t idx = (((size_t)b_ * H_ + h) * T_ + t) * DK_ + dk;
      out0[idx] = (__bf16)(val + bu[ncol]);
      out1[idx] = (__bf16)(val + bvp[ncol]);
    } else if (MODE == 1) {
      int b_ = m >> 11, t = m & (T_ - 1);
      out0[(((size_t)b_ * H_ + h) * T_ + t) * DK_ + dk] = (__bf16)val;
    } else if (MODE == 2) {
      int b_ = m >> 11, t = m & (T_ - 1);
      out0[(((size_t)b_ * H_ + h) * DK_ + dk) * T_ + t] = (__bf16)val;   // (B,H,DK,T)
    } else {
      out0[((size_t)h * PPAD_ + m) * DK_ + dk] = (__bf16)val;            // (H,PPAD,DK)
    }
  }
}

// ---------------------------------------------------------------------------
// Scores: S[bh,t,j] = ( q_u[t]·k[j] + q_v[t]·p[j-t+T-1] ) / 8
// One wave -> 16 (t) x 64 (j). Banded bd: the 64-wide j block needs p rows
// [r0, r0+80) = 5 WMMA tiles shared by the 4 sub-tiles; staged in LDS and
// gathered with the analytic rel_shift  c = 16*i + n - m + 15.
// ---------------------------------------------------------------------------
__global__ void scores_kernel(const __bf16* __restrict__ qu, const __bf16* __restrict__ qv,
                              const __bf16* __restrict__ kh, const __bf16* __restrict__ ph,
                              float* __restrict__ S) {
  int bh = blockIdx.z;
  int t0 = blockIdx.y * 16;
  int j0 = blockIdx.x * 64;
  const __bf16* quB = qu + (size_t)bh * T_ * DK_;
  const __bf16* qvB = qv + (size_t)bh * T_ * DK_;
  const __bf16* khB = kh + (size_t)bh * T_ * DK_;
  const __bf16* phB = ph + (size_t)(bh & (H_ - 1)) * PPAD_ * DK_;
  int r0 = j0 - t0 + (T_ - 16);          // >= 0; r0+79 <= 4095 < PPAD
  v8f ac[4] = {};
  v8f g[5]  = {};
#pragma unroll
  for (int k0 = 0; k0 < DK_; k0 += 32) {
    v16bf a_u = load_a_bf16(quB, DK_, t0, k0);
    v16bf a_v = load_a_bf16(qvB, DK_, t0, k0);
#pragma unroll
    for (int i = 0; i < 4; ++i)
      ac[i] = wmma_bf16(a_u, load_b_bf16(khB, DK_, j0 + 16 * i, k0), ac[i]);
#pragma unroll
    for (int gi = 0; gi < 5; ++gi)
      g[gi] = wmma_bf16(a_v, load_b_bf16(phB, DK_, r0 + 16 * gi, k0), g[gi]);
  }
  __shared__ float bdl[16][80];
  int lane = threadIdx.x & 31, half = lane >> 4, n = lane & 15;
#pragma unroll
  for (int gi = 0; gi < 5; ++gi)
#pragma unroll
    for (int r = 0; r < 8; ++r) {
      int m = r + half * 8;
      bdl[m][16 * gi + n] = g[gi][r];   // p row r0 + 16*gi + n
    }
  __syncthreads();
  float* Srow = S + ((size_t)bh * T_ + t0) * T_ + j0;
#pragma unroll
  for (int i = 0; i < 4; ++i)
#pragma unroll
    for (int r = 0; r < 8; ++r) {
      int m = r + half * 8;                 // dt
      int c = 16 * i + n - m + 15;          // rel-shift gather, 0..78
      Srow[(size_t)m * T_ + 16 * i + n] = (ac[i][r] + bdl[m][c]) * 0.125f;
    }
}

// ---------------------------------------------------------------------------
// Masked softmax; writes bf16 attn IN PLACE over the row's own fp32 storage
// (row fully read into registers first; one block owns one row).
// ---------------------------------------------------------------------------
__global__ void softmax_kernel(float* __restrict__ S, const int* __restrict__ mask) {
  int row = blockIdx.x;                 // bh*T + t
  int b = row / (H_ * T_);
  float* srow = S + (size_t)row * T_;
  const int* mrow = mask + (size_t)b * T_;
  int tid = threadIdx.x;
  float v[8]; int mk[8];
  float mx = -3.0e38f;
#pragma unroll
  for (int i = 0; i < 8; ++i) {
    int idx = tid + i * 256;
    float s = srow[idx];
    int m = mrow[idx];
    s = (m == 0) ? -1.0e9f : s;
    v[i] = s; mk[i] = m;
    mx = fmaxf(mx, s);
  }
  __shared__ float red[256];
  red[tid] = mx; __syncthreads();
  for (int s = 128; s > 0; s >>= 1) { if (tid < s) red[tid] = fmaxf(red[tid], red[tid + s]); __syncthreads(); }
  float rowmax = red[0]; __syncthreads();
  float sum = 0.0f;
#pragma unroll
  for (int i = 0; i < 8; ++i) { v[i] = __expf(v[i] - rowmax); sum += v[i]; }
  red[tid] = sum; __syncthreads();
  for (int s = 128; s > 0; s >>= 1) { if (tid < s) red[tid] += red[tid + s]; __syncthreads(); }
  float inv = 1.0f / red[0];
  __bf16* arow = (__bf16*)srow;
#pragma unroll
  for (int i = 0; i < 8; ++i) {
    int idx = tid + i * 256;
    float a = v[i] * inv;
    if (mk[i] == 0) a = 0.0f;
    arow[idx] = (__bf16)a;
  }
}

// ---------------------------------------------------------------------------
// attn @ v. Block = 4 waves = 64 query rows; each wave -> 16 x DK(=64) output.
// The shared v-operand tile (64 rows x 32 k, bf16, 4 KB) is staged into LDS
// by the Tensor Data Mover (tensor_load_to_lds + s_wait_tensorcnt), then all
// waves consume it via ds_load-fed B fragments. attn is the bf16 overlay with
// row pitch 2T elements.
// ---------------------------------------------------------------------------
__global__ void attnv_kernel(const __bf16* __restrict__ attn, const __bf16* __restrict__ vT,
                             __bf16* __restrict__ X) {
  int bh = blockIdx.y;
  int wave = threadIdx.x >> 5;
  int t0 = blockIdx.x * 64 + wave * 16;
  const __bf16* aB = attn + (size_t)bh * T_ * (2 * T_);
  const __bf16* vB = vT + (size_t)bh * DK_ * T_;
  __shared__ __bf16 vstage[DK_][32];
  const unsigned lds_addr = (unsigned)(size_t)(&vstage[0][0]);   // LDS byte offset
  v8f acc[4] = {};
  for (int k0 = 0; k0 < T_; k0 += 32) {
    if (wave == 0) {
      // Tensor DMA descriptor (D#): 2-D tile, 2-byte elems, 32 x 64, row pitch T
      unsigned long long ga = (unsigned long long)(size_t)(vB + k0);
      v4u g0;
      g0[0] = 1u;                                             // count=1
      g0[1] = lds_addr;                                       // lds_addr
      g0[2] = (unsigned)(ga & 0xffffffffu);                   // global_addr[31:0]
      g0[3] = (unsigned)((ga >> 32) & 0x1ffffffu) | (2u << 30); // addr[56:32] | type=2
      v8i g1;
      g1[0] = (int)(1u << 16);                 // data_size=1 (2 bytes)
      g1[1] = (int)((unsigned)T_ << 16);       // tensor_dim0 = T (bits 79:48)
      g1[2] = (int)((unsigned)DK_ << 16);      // tensor_dim1 = 64 (bits 111:80)
      g1[3] = (int)(32u << 16);                // tile_dim0 = 32 (bits 127:112)
      g1[4] = DK_;                             // tile_dim1 = 64; tile_dim2 = 0
      g1[5] = T_;                              // tensor_dim0_stride = T
      g1[6] = 0;
      g1[7] = 0;
      v4i z4 = {0, 0, 0, 0};
#if defined(__clang_major__) && (__clang_major__ >= 23)
      v8i z8 = {0, 0, 0, 0, 0, 0, 0, 0};
      __builtin_amdgcn_tensor_load_to_lds(g0, g1, z4, z4, z8, 0);
#else
      __builtin_amdgcn_tensor_load_to_lds(g0, g1, z4, z4, 0);
#endif
      __builtin_amdgcn_s_wait_tensorcnt(0);
    }
    __syncthreads();
    if (k0 + 32 < T_)
      __builtin_prefetch(aB + (size_t)t0 * (2 * T_) + k0 + 32, 0, 1);
    v16bf a = load_a_bf16(aB, 2 * T_, t0, k0);
    const __bf16* vs = &vstage[0][0];
#pragma unroll
    for (int i = 0; i < 4; ++i)
      acc[i] = wmma_bf16(a, load_b_bf16(vs, 32, 16 * i, 0), acc[i]);
    __syncthreads();
  }
  int lane = threadIdx.x & 31, half = lane >> 4, n = lane & 15;
  int b = bh >> 2, h = bh & (H_ - 1);
#pragma unroll
  for (int r = 0; r < 8; ++r) {
    int m = r + half * 8;
    size_t base = ((size_t)b * T_ + t0 + m) * D_ + h * DK_;
#pragma unroll
    for (int i = 0; i < 4; ++i)
      X[base + 16 * i + n] = (__bf16)acc[i][r];
  }
}

// ---------------------------------------------------------------------------
// Output projection: out(fp32) = X(bf16, B*T x 256) @ Wo^T(bf16) + bo
// ---------------------------------------------------------------------------
__global__ void out_gemm_kernel(const __bf16* __restrict__ X, const __bf16* __restrict__ Wo,
                                const float* __restrict__ bo, float* __restrict__ out) {
  int n0 = blockIdx.x * 16;
  int m0 = blockIdx.y * 16;
  v8f acc = {};
#pragma unroll
  for (int k0 = 0; k0 < D_; k0 += 32) {
    v16bf a = load_a_bf16(X, D_, m0, k0);
    v16bf b = load_b_bf16(Wo, D_, n0, k0);
    acc = wmma_bf16(a, b, acc);
  }
  int lane = threadIdx.x & 31, half = lane >> 4, n = n0 + (lane & 15);
#pragma unroll
  for (int r = 0; r < 8; ++r) {
    int m = m0 + r + half * 8;
    out[(size_t)m * D_ + n] = acc[r] + bo[n];
  }
}

// ---------------------------------------------------------------------------
// Launch. Workspace requirement ~290 MB (scores fp32 dominates; attn bf16 is
// overlaid in place).
// ---------------------------------------------------------------------------
extern "C" void kernel_launch(void* const* d_in, const int* in_sizes, int n_in,
                              void* d_out, int out_size, void* d_ws, size_t ws_size,
                              hipStream_t stream) {
  (void)in_sizes; (void)n_in; (void)out_size; (void)ws_size;
  const float* query   = (const float*)d_in[0];
  const float* key     = (const float*)d_in[1];
  const float* value   = (const float*)d_in[2];
  const float* pos_emb = (const float*)d_in[3];
  const int*   mask    = (const int*)d_in[4];
  const float* Wq  = (const float*)d_in[5];
  const float* bq  = (const float*)d_in[6];
  const float* Wk  = (const float*)d_in[7];
  const float* bk  = (const float*)d_in[8];
  const float* Wv  = (const float*)d_in[9];
  const float* bv  = (const float*)d_in[10];
  const float* Wpos= (const float*)d_in[11];
  const float* pbu = (const float*)d_in[12];
  const float* pbv = (const float*)d_in[13];
  const float* Wo  = (const float*)d_in[14];
  const float* bo  = (const float*)d_in[15];

  char* ws = (char*)d_ws;
  size_t off = 0;
  auto alloc = [&](size_t bytes) -> void* {
    void* p = ws + off;
    off = (off + bytes + 255) & ~(size_t)255;
    return p;
  };
  const int NW = D_ * D_;
  __bf16* wq_b = (__bf16*)alloc(NW * sizeof(__bf16));
  __bf16* wk_b = (__bf16*)alloc(NW * sizeof(__bf16));
  __bf16* wv_b = (__bf16*)alloc(NW * sizeof(__bf16));
  __bf16* wp_b = (__bf16*)alloc(NW * sizeof(__bf16));
  __bf16* wo_b = (__bf16*)alloc(NW * sizeof(__bf16));
  const size_t qkN = (size_t)B_ * H_ * T_ * DK_;
  __bf16* qu = (__bf16*)alloc(qkN * sizeof(__bf16));
  __bf16* qv = (__bf16*)alloc(qkN * sizeof(__bf16));
  __bf16* kh = (__bf16*)alloc(qkN * sizeof(__bf16));
  __bf16* vT = (__bf16*)alloc(qkN * sizeof(__bf16));
  const size_t phN = (size_t)H_ * PPAD_ * DK_;
  __bf16* ph = (__bf16*)alloc(phN * sizeof(__bf16));
  float*  S  = (float*)alloc((size_t)B_ * H_ * T_ * T_ * sizeof(float));
  __bf16* xh = (__bf16*)alloc((size_t)B_ * T_ * D_ * sizeof(__bf16));

  // 1) weights fp32 -> bf16; zero p pad
  int cb = (NW + 255) / 256;
  cvt_f32_bf16_kernel<<<cb, 256, 0, stream>>>(Wq,   wq_b, NW);
  cvt_f32_bf16_kernel<<<cb, 256, 0, stream>>>(Wk,   wk_b, NW);
  cvt_f32_bf16_kernel<<<cb, 256, 0, stream>>>(Wv,   wv_b, NW);
  cvt_f32_bf16_kernel<<<cb, 256, 0, stream>>>(Wpos, wp_b, NW);
  cvt_f32_bf16_kernel<<<cb, 256, 0, stream>>>(Wo,   wo_b, NW);
  zero_bf16_kernel<<<(int)((phN + 255) / 256), 256, 0, stream>>>(ph, (int)phN);

  // 2) projections (WMMA)
  dim3 pgrid(D_ / 16, (B_ * T_) / 16);
  proj_gemm_kernel<0, false><<<pgrid, 32, 0, stream>>>(query, wq_b, bq, pbu, pbv, qu, qv, B_ * T_);
  proj_gemm_kernel<1, false><<<pgrid, 32, 0, stream>>>(key,   wk_b, bk, nullptr, nullptr, kh, nullptr, B_ * T_);
  proj_gemm_kernel<2, false><<<pgrid, 32, 0, stream>>>(value, wv_b, bv, nullptr, nullptr, vT, nullptr, B_ * T_);
  proj_gemm_kernel<3, true><<<dim3(D_ / 16, (P_ + 15) / 16), 32, 0, stream>>>(
      pos_emb, wp_b, nullptr, nullptr, nullptr, ph, nullptr, P_);

  // 3) fused ac + banded-bd (analytic rel_shift) + scale (WMMA, 16x64/wave)
  scores_kernel<<<dim3(T_ / 64, T_ / 16, B_ * H_), 32, 0, stream>>>(qu, qv, kh, ph, S);

  // 4) masked softmax, bf16 attn overlaid in place
  softmax_kernel<<<B_ * H_ * T_, 256, 0, stream>>>(S, mask);

  // 5) attn @ v  (WMMA + TDM-staged v tiles), direct (B,T,D) interleave
  attnv_kernel<<<dim3(T_ / 64, B_ * H_), 128, 0, stream>>>((const __bf16*)S, vT, xh);

  // 6) output projection (WMMA), fp32 out
  out_gemm_kernel<<<dim3(D_ / 16, (B_ * T_) / 16), 32, 0, stream>>>(xh, wo_b, bo, (float*)d_out);
}
